// ActorCritic_70016556859504
// MI455X (gfx1250) — matrix operand order repaired and checked by reading, not compiled, not run on hardware
//
#include <hip/hip_runtime.h>
#include <hip/hip_bf16.h>

typedef __bf16 bf16;
typedef __attribute__((ext_vector_type(16))) __bf16 v16bf;
typedef __attribute__((ext_vector_type(8)))  float  v8f;
typedef __attribute__((ext_vector_type(4)))  int    v4i_t;

// pointer-to-AS1 v4i (global) and pointer-to-AS3 v4i (LDS)
typedef __attribute__((address_space(1))) v4i_t* gv4i_p;
typedef __attribute__((address_space(3))) v4i_t* lv4i_p;

#if __has_builtin(__builtin_amdgcn_global_load_async_to_lds_b128) && \
    __has_builtin(__builtin_amdgcn_s_wait_asynccnt)
#define USE_ASYNC_LDS 1
#else
#define USE_ASYNC_LDS 0
#endif

// ---------------------------------------------------------------------------
// Async (ASYNCcnt-tracked) global -> LDS staging; uint4 fallback if the
// builtin isn't declared. bytes must be a multiple of 16, pointers 16B-aligned.
// ---------------------------------------------------------------------------
__device__ __forceinline__ void lds_copy_issue(void* lds, const void* gmem,
                                               int bytes, int tid, int nth) {
#if USE_ASYNC_LDS
    for (int off = tid * 16; off < bytes; off += nth * 16)
        __builtin_amdgcn_global_load_async_to_lds_b128(
            (gv4i_p)((const char*)gmem + off),
            (lv4i_p)((char*)lds + off), 0, 0);
#else
    for (int off = tid * 16; off < bytes; off += nth * 16)
        *(uint4*)((char*)lds + off) = *(const uint4*)((const char*)gmem + off);
#endif
}
__device__ __forceinline__ void lds_copy_wait() {
#if USE_ASYNC_LDS
    __builtin_amdgcn_s_wait_asynccnt(0);
#endif
}

// ---------------------------------------------------------------------------
// WMMA fragment helpers (CDNA5 wave32 layouts, CDNA5 ISA §7.12.2)
// ---------------------------------------------------------------------------
__device__ __forceinline__ v16bf load_a_frag(const bf16* p, int ld, int lane) {
    v16bf f;
    const int m  = lane & 15;
    const int hi = (lane >> 4) & 1;
    const bf16* row = p + m * ld;
#pragma unroll
    for (int v = 0; v < 8; ++v) {
        int k0 = hi * 16 + 2 * v;
        f[2 * v]     = row[k0];
        f[2 * v + 1] = row[k0 + 1];
    }
    return f;
}

__device__ __forceinline__ v16bf load_b_frag(const bf16* p, int ld, int lane) {
    v16bf f;
    const int n  = lane & 15;
    const int hi = (lane >> 4) & 1;
#pragma unroll
    for (int v = 0; v < 8; ++v) {
        int k0 = hi * 16 + 2 * v;
        f[2 * v]     = p[k0 * ld + n];
        f[2 * v + 1] = p[(k0 + 1) * ld + n];
    }
    return f;
}

__device__ __forceinline__ void store_c_frag(float* p, int ld, int lane, v8f c) {
    const int n     = lane & 15;
    const int mbase = ((lane >> 4) & 1) * 8;
#pragma unroll
    for (int v = 0; v < 8; ++v) p[(mbase + v) * ld + n] = c[v];
}

__device__ __forceinline__ float lrelu(float x, float s) { return x > 0.f ? x : s * x; }
__device__ __forceinline__ float gelu_exact(float x) {
    return 0.5f * x * (1.0f + erff(x * 0.70710678118654752f));
}

// ---------------------------------------------------------------------------
// One-time f32 -> bf16 weight conversion (weights then stay bf16 in L2).
// ---------------------------------------------------------------------------
__global__ void cvt_kernel(const float* __restrict__ src, bf16* __restrict__ dst,
                           int nelem) {
    for (int i = blockIdx.x * blockDim.x + threadIdx.x; i < nelem;
         i += gridDim.x * blockDim.x)
        dst[i] = (bf16)src[i];
}

// ---------------------------------------------------------------------------
// Kernel 1: GAT layer 1 (heads=2, ch=128, concat) fused per graph.
// ---------------------------------------------------------------------------
__global__ void gat1_kernel(const float* __restrict__ state,
                            const bf16* __restrict__ W1b,
                            const float* __restrict__ as1,
                            const float* __restrict__ ad1,
                            const float* __restrict__ b1,
                            bf16* __restrict__ x1) {
    extern __shared__ char smem[];
    float* sXH = (float*)smem;                          // 64x256 f32   (64 KB)
    bf16*  sX  = (bf16*)(smem + 65536);                 // 64x64 bf16   ( 8 KB)
    bf16*  sW  = (bf16*)(smem + 65536 + 8192);          // 64x256 bf16  (32 KB)
    float* sES = (float*)(smem + 65536 + 8192 + 32768); // 50x2
    float* sED = sES + 128;

    const int b = blockIdx.x, tid = threadIdx.x;
    const int lane = tid & 31, wave = tid >> 5;

    // weights: pure bf16 copy -> async DMA to LDS
    lds_copy_issue(sW, W1b, 64 * 256 * 2, tid, blockDim.x);
    // activations need f32->bf16 conversion (small: 64x64)
    for (int idx = tid; idx < 64 * 64; idx += blockDim.x) {
        int r = idx >> 6, c = idx & 63;
        float v = (r < 50) ? state[(b * 50 + r) * 64 + c] : 0.f;
        sX[idx] = (bf16)v;
    }
    lds_copy_wait();
    __syncthreads();

    // GEMM: M=64(4 tiles) x N=256(4 tiles/wave) x K=64(2 steps of 32)
    for (int m = 0; m < 4; ++m) {
        for (int nt = 0; nt < 4; ++nt) {
            const int ncol = wave * 64 + nt * 16;
            v8f c = {};
#pragma unroll
            for (int k = 0; k < 2; ++k) {
                v16bf a  = load_a_frag(sX + m * 16 * 64 + k * 32, 64, lane);
                v16bf bb = load_b_frag(sW + k * 32 * 256 + ncol, 256, lane);
                c = __builtin_amdgcn_wmma_f32_16x16x32_bf16(false, a, false, bb,
                                                            (short)0, c, false, false);
            }
            store_c_frag(sXH + m * 16 * 256 + ncol, 256, lane, c);
        }
    }
    __syncthreads();

    if (tid < 100) {
        int node = tid >> 1, h = tid & 1;
        const float* row = sXH + node * 256 + h * 128;
        float es = 0.f, ed = 0.f;
        for (int c = 0; c < 128; ++c) {
            float v = row[c];
            es += v * as1[h * 128 + c];
            ed += v * ad1[h * 128 + c];
        }
        sES[node * 2 + h] = es;
        sED[node * 2 + h] = ed;
    }
    __syncthreads();

    for (int ch = tid; ch < 256; ch += blockDim.x) {
        const int h = ch >> 7;
        for (int i = 0; i < 50; ++i) {
            int js[3]; int cnt = 0;
            js[cnt++] = i;
            if (i > 0)  js[cnt++] = i - 1;
            if (i < 49) js[cnt++] = i + 1;
            const float edi = sED[i * 2 + h];
            float e[3], mx = -1e30f;
            for (int t = 0; t < cnt; ++t) {
                e[t] = lrelu(sES[js[t] * 2 + h] + edi, 0.2f);
                mx = fmaxf(mx, e[t]);
            }
            float den = 0.f, acc = 0.f;
            for (int t = 0; t < cnt; ++t) {
                float w = __expf(e[t] - mx);
                den += w;
                acc += w * sXH[js[t] * 256 + ch];
            }
            float o = gelu_exact(acc / den + b1[ch]);
            x1[(b * 50 + i) * 256 + ch] = (bf16)o;
        }
    }
}

// ---------------------------------------------------------------------------
// Kernel 2: GAT layer 2 (heads=1, ch=128) + graph mean pool.
// ---------------------------------------------------------------------------
__global__ void gat2_kernel(const bf16* __restrict__ x1,
                            const bf16* __restrict__ W2b,
                            const float* __restrict__ as2,
                            const float* __restrict__ ad2,
                            const float* __restrict__ b2,
                            bf16* __restrict__ x2,
                            bf16* __restrict__ gemb) {
    extern __shared__ char smem[];
    float* sXH = (float*)smem;                 // 64x128 f32   (32 KB)
    bf16*  sX  = (bf16*)(smem + 32768);        // 64x256 bf16  (32 KB)
    bf16*  sW  = (bf16*)(smem + 65536);        // 256x128 bf16 (64 KB)
    float* sES = (float*)(smem + 131072);      // 50
    float* sED = sES + 64;

    const int b = blockIdx.x, tid = threadIdx.x;
    const int lane = tid & 31, wave = tid >> 5;

    // both inputs are already bf16 -> async DMA to LDS
    lds_copy_issue(sX, x1 + (size_t)b * 50 * 256, 50 * 256 * 2, tid, blockDim.x);
    lds_copy_issue(sW, W2b, 256 * 128 * 2, tid, blockDim.x);
    for (int idx = 50 * 256 + tid; idx < 64 * 256; idx += blockDim.x)
        sX[idx] = (bf16)0.0f;                  // zero-pad rows 50..63
    lds_copy_wait();
    __syncthreads();

    // GEMM: M=64(4) x N=128(2 tiles/wave) x K=256(8 steps)
    for (int m = 0; m < 4; ++m) {
        for (int nt = 0; nt < 2; ++nt) {
            const int ncol = wave * 32 + nt * 16;
            v8f c = {};
#pragma unroll
            for (int k = 0; k < 8; ++k) {
                v16bf a  = load_a_frag(sX + m * 16 * 256 + k * 32, 256, lane);
                v16bf bb = load_b_frag(sW + k * 32 * 128 + ncol, 128, lane);
                c = __builtin_amdgcn_wmma_f32_16x16x32_bf16(false, a, false, bb,
                                                            (short)0, c, false, false);
            }
            store_c_frag(sXH + m * 16 * 128 + ncol, 128, lane, c);
        }
    }
    __syncthreads();

    if (tid < 50) {
        const float* row = sXH + tid * 128;
        float es = 0.f, ed = 0.f;
        for (int c = 0; c < 128; ++c) {
            float v = row[c];
            es += v * as2[c];
            ed += v * ad2[c];
        }
        sES[tid] = es;
        sED[tid] = ed;
    }
    __syncthreads();

    if (tid < 128) {
        const int ch = tid;
        float gsum = 0.f;
        for (int i = 0; i < 50; ++i) {
            int js[3]; int cnt = 0;
            js[cnt++] = i;
            if (i > 0)  js[cnt++] = i - 1;
            if (i < 49) js[cnt++] = i + 1;
            const float edi = sED[i];
            float e[3], mx = -1e30f;
            for (int t = 0; t < cnt; ++t) {
                e[t] = lrelu(sES[js[t]] + edi, 0.2f);
                mx = fmaxf(mx, e[t]);
            }
            float den = 0.f, acc = 0.f;
            for (int t = 0; t < cnt; ++t) {
                float w = __expf(e[t] - mx);
                den += w;
                acc += w * sXH[js[t] * 128 + ch];
            }
            float o = gelu_exact(acc / den + b2[ch]);
            x2[(b * 50 + i) * 128 + ch] = (bf16)o;
            gsum += o;
        }
        gemb[b * 128 + ch] = (bf16)(gsum * (1.0f / 50.0f));
    }
}

// ---------------------------------------------------------------------------
// Kernel 3/4: 3-layer MLP, 64-row tiles; bf16 weights staged via async DMA.
// ---------------------------------------------------------------------------
__global__ void mlp_kernel(const bf16* __restrict__ in,
                           const bf16* __restrict__ w1b, const float* __restrict__ bb1,
                           const bf16* __restrict__ w2b, const float* __restrict__ bb2,
                           const float* __restrict__ w3, const float* __restrict__ bb3,
                           int nout, float* __restrict__ out) {
    extern __shared__ char smem[];
    float* sH  = (float*)smem;             // 64x128 f32   (32 KB)
    bf16*  sIn = (bf16*)(smem + 32768);    // 64x128 bf16  (16 KB)
    bf16*  sW  = (bf16*)(smem + 49152);    // 128x128 bf16 (32 KB)

    const int r0 = blockIdx.x * 64, tid = threadIdx.x;
    const int lane = tid & 31, wave = tid >> 5;

    lds_copy_issue(sIn, in + (size_t)r0 * 128, 64 * 128 * 2, tid, blockDim.x);
    lds_copy_issue(sW, w1b, 128 * 128 * 2, tid, blockDim.x);
    lds_copy_wait();
    __syncthreads();

#pragma unroll
    for (int layer = 0; layer < 2; ++layer) {
        // GEMM: M=64(4) x N=128(2 tiles/wave) x K=128(4 steps)
        for (int m = 0; m < 4; ++m) {
            for (int nt = 0; nt < 2; ++nt) {
                const int ncol = wave * 32 + nt * 16;
                v8f c = {};
#pragma unroll
                for (int k = 0; k < 4; ++k) {
                    v16bf a  = load_a_frag(sIn + m * 16 * 128 + k * 32, 128, lane);
                    v16bf bb = load_b_frag(sW + k * 32 * 128 + ncol, 128, lane);
                    c = __builtin_amdgcn_wmma_f32_16x16x32_bf16(false, a, false, bb,
                                                                (short)0, c, false, false);
                }
                store_c_frag(sH + m * 16 * 128 + ncol, 128, lane, c);
            }
        }
        __syncthreads();
        // overlap: kick off next-layer weight DMA, then requantize activations
        if (layer == 0)
            lds_copy_issue(sW, w2b, 128 * 128 * 2, tid, blockDim.x);
        const float* bia = (layer == 0) ? bb1 : bb2;
        for (int idx = tid; idx < 64 * 128; idx += blockDim.x)
            sIn[idx] = (bf16)lrelu(sH[idx] + bia[idx & 127], 0.01f);
        lds_copy_wait();
        __syncthreads();
    }

    // final narrow layer: 128 -> nout (4 or 1), scalar dot products
    for (int idx = tid; idx < 64 * nout; idx += blockDim.x) {
        const int row = idx / nout, col = idx % nout;
        float acc = 0.f;
        for (int k = 0; k < 128; ++k)
            acc += (float)sIn[row * 128 + k] * w3[k * nout + col];
        out[(size_t)(r0 + row) * nout + col] = acc + bb3[col];
    }
}

// ---------------------------------------------------------------------------
extern "C" void kernel_launch(void* const* d_in, const int* in_sizes, int n_in,
                              void* d_out, int out_size, void* d_ws, size_t ws_size,
                              hipStream_t stream) {
    const float* state = (const float*)d_in[0];
    const float* W1    = (const float*)d_in[1];
    const float* as1   = (const float*)d_in[2];
    const float* ad1   = (const float*)d_in[3];
    const float* b1    = (const float*)d_in[4];
    const float* W2    = (const float*)d_in[5];
    const float* as2   = (const float*)d_in[6];
    const float* ad2   = (const float*)d_in[7];
    const float* b2    = (const float*)d_in[8];
    const float* aw1   = (const float*)d_in[9];
    const float* ab1   = (const float*)d_in[10];
    const float* aw2   = (const float*)d_in[11];
    const float* ab2   = (const float*)d_in[12];
    const float* aw3   = (const float*)d_in[13];
    const float* ab3   = (const float*)d_in[14];
    const float* vw1   = (const float*)d_in[15];
    const float* vb1   = (const float*)d_in[16];
    const float* vw2   = (const float*)d_in[17];
    const float* vb2   = (const float*)d_in[18];
    const float* vw3   = (const float*)d_in[19];
    const float* vb3   = (const float*)d_in[20];

    const int B = 2048, n = 50;

    // workspace carve (byte offsets, all 16B aligned)
    char* ws = (char*)d_ws;
    bf16* x1    = (bf16*)ws;                                   // B*n*256 bf16
    bf16* x2    = (bf16*)(ws + (size_t)B * n * 256 * 2);       // B*n*128 bf16
    bf16* gemb  = (bf16*)(ws + (size_t)B * n * 384 * 2);       // B*128  bf16
    char* wb    = ws + (size_t)B * n * 384 * 2 + (size_t)B * 128 * 2;
    bf16* wW1   = (bf16*)(wb);               // 64*256
    bf16* wW2   = (bf16*)(wb + 32768);       // 256*128
    bf16* waw1  = (bf16*)(wb + 98304);       // 128*128
    bf16* waw2  = (bf16*)(wb + 131072);      // 128*128
    bf16* wvw1  = (bf16*)(wb + 163840);      // 128*128
    bf16* wvw2  = (bf16*)(wb + 196608);      // 128*128

    float* out_mean  = (float*)d_out;                 // [B*n, 4] flat
    float* out_value = (float*)d_out + (size_t)B * n * 4;

    // one-time weight precision conversion (stays resident in L2)
    cvt_kernel<<<64, 256, 0, stream>>>(W1,  wW1,  64 * 256);
    cvt_kernel<<<64, 256, 0, stream>>>(W2,  wW2,  256 * 128);
    cvt_kernel<<<64, 256, 0, stream>>>(aw1, waw1, 128 * 128);
    cvt_kernel<<<64, 256, 0, stream>>>(aw2, waw2, 128 * 128);
    cvt_kernel<<<64, 256, 0, stream>>>(vw1, wvw1, 128 * 128);
    cvt_kernel<<<64, 256, 0, stream>>>(vw2, wvw2, 128 * 128);

    const size_t shm1 = 65536 + 8192 + 32768 + 1024;   // 107,520 B
    const size_t shm2 = 32768 + 32768 + 65536 + 1024;  // 132,096 B
    const size_t shm3 = 32768 + 16384 + 32768;         //  81,920 B

    gat1_kernel<<<B, 128, shm1, stream>>>(state, wW1, as1, ad1, b1, x1);
    gat2_kernel<<<B, 128, shm2, stream>>>(x1, wW2, as2, ad2, b2, x2, gemb);
    mlp_kernel<<<(B * n) / 64, 128, shm3, stream>>>(x2, waw1, ab1, waw2, ab2,
                                                    aw3, ab3, 4, out_mean);
    mlp_kernel<<<B / 64, 128, shm3, stream>>>(gemb, wvw1, vb1, wvw2, vb2,
                                              vw3, vb3, 1, out_value);
}
